// RNNLMScratch_46205258170721
// MI455X (gfx1250) — compile-verified, matrix operand order checked by reading, not run on hardware
//
#include <hip/hip_runtime.h>
#include <math.h>

typedef __attribute__((ext_vector_type(16))) _Float16 v16h;
typedef __attribute__((ext_vector_type(8)))  _Float16 v8h;
typedef __attribute__((ext_vector_type(8)))  float    v8f;

#define B_   1024
#define T_   256
#define V_   256
#define H_   32
#define BG_  (B_/16)          // 64 batch tiles of 16 rows
#define HACC_STRIDE 36        // pad: all lanes hit distinct bank granules; 16B aligned

// Branch-free tanh: prefer the CDNA5 V_TANH_F32 TRANS op if the builtin exists;
// otherwise exp2 + fast rcp (branch-free, keeps the row-phase loop vectorizable).
__device__ __forceinline__ float fast_tanhf(float x) {
#if __has_builtin(__builtin_amdgcn_tanhf)
    return __builtin_amdgcn_tanhf(x);
#elif __has_builtin(__builtin_amdgcn_tanh_f32)
    return __builtin_amdgcn_tanh_f32(x);
#else
    // tanh(x) = (e^{2x} - 1) / (e^{2x} + 1);  2*log2(e) = 2.88539008...
    const float e = __builtin_amdgcn_exp2f(x * 2.8853900817779268f);
    return (e - 1.0f) * __builtin_amdgcn_rcpf(e + 1.0f);
#endif
}

// ---------------------------------------------------------------------------
// Kernel 1: recurrent scan. One wave per 16-batch tile. h lives as a WMMA
// A-fragment in registers; two v_wmma_f32_16x16x32_f16 per step compute
// h @ W_hh. The D->A transpose goes through one LDS round-trip, but each lane
// recomputes exactly the 16 columns its own A-fragment needs, so the new
// fragment is assembled directly in registers (no second LDS trip).
// ---------------------------------------------------------------------------
__global__ __launch_bounds__(32)
void rnn_scan_kernel(const int*   __restrict__ X,      // [B, T]
                     const float* __restrict__ W_xh,   // [V, H]
                     const float* __restrict__ W_hh,   // [H, H]
                     const float* __restrict__ b_h,    // [H]
                     _Float16*    __restrict__ hsFrag) // [BG*T][32 lanes][16 halfs]
{
    __shared__ __align__(16) float lds_wxh[V_ * H_];        // W_xh + b_h, 32 KB
    __shared__ __align__(16) float hacc[16 * HACC_STRIDE];  // f32 h@W_hh tile

    const int lane = threadIdx.x;                  // 0..31 (wave32)
    const int bg   = blockIdx.x;                   // batch tile

    // Preload W_xh into LDS with b_h folded in (saves 16 adds per step).
    {
        const float4* src = (const float4*)W_xh;
        const float4* bh4 = (const float4*)b_h;    // 8 float4 covering H=32
        float4*       dst = (float4*)lds_wxh;
        #pragma unroll
        for (int i = lane; i < (V_ * H_) / 4; i += 32) {
            float4 v = src[i];
            float4 b = bh4[i & 7];
            v.x += b.x; v.y += b.y; v.z += b.z; v.w += b.w;
            dst[i] = v;
        }
    }

    // Resident B-fragments of W_hh (two 32x16 tiles, f16).
    // B layout: lanes 0-15 -> col N=lane, halfs = K{0..7,16..23};
    //           lanes 16-31 -> col N=lane-16, halfs = K{8..15,24..31}.
    const int colL = lane & 15;
    const int kb   = (lane < 16) ? 0 : 8;
    v16h bhh0, bhh1;
    #pragma unroll
    for (int i = 0; i < 8; ++i) {
        const int k1 = kb + i;
        const int k2 = kb + 16 + i;
        bhh0[i]     = (_Float16)W_hh[k1 * H_ + colL];
        bhh0[i + 8] = (_Float16)W_hh[k2 * H_ + colL];
        bhh1[i]     = (_Float16)W_hh[k1 * H_ + colL + 16];
        bhh1[i + 8] = (_Float16)W_hh[k2 * H_ + colL + 16];
    }

    // Row-phase mapping: lane computes row = lane&15, cols {kb..kb+7, 16+kb..16+kb+7}
    // == exactly the values of its own next A-fragment.
    const int row = lane & 15;
    const int rb  = (lane < 16) ? 0 : 8;           // D-fragment row base
    const int c0i = kb >> 2;                       // float4 index of col chunk kb
    const int c1i = 4 + (kb >> 2);                 // float4 index of col chunk 16+kb

    const int* xrow = X + (bg * 16 + row) * T_;
    int tok_next = xrow[0];                        // software-pipelined token

    v16h a = {};                                   // h(0) == 0
    __syncthreads();

    for (int t = 0; t < T_; ++t) {
        const int tok = tok_next;
        tok_next = xrow[(t + 1 < T_) ? t + 1 : t]; // prefetch next step's token

        // Embedding row chunks (bias pre-folded); independent of the WMMAs,
        // so these ds loads overlap the matrix ops.
        const float4* wer = (const float4*)&lds_wxh[tok * H_];
        const float4 we0 = wer[c0i],     we1 = wer[c0i + 1];
        const float4 we2 = wer[c1i],     we3 = wer[c1i + 1];

        // h @ W_hh (zero accumulator; at t=0 A==0 so the result is 0, correct).
        v8f z = {};
        v8f d0 = __builtin_amdgcn_wmma_f32_16x16x32_f16(false, a, false, bhh0,
                                                        (short)0, z, false, false);
        v8f d1 = __builtin_amdgcn_wmma_f32_16x16x32_f16(false, a, false, bhh1,
                                                        (short)0, z, false, false);

        // Scatter D fragments to row-major LDS (transpose across lanes).
        #pragma unroll
        for (int r = 0; r < 8; ++r) {
            hacc[(rb + r) * HACC_STRIDE + colL]      = d0[r];
            hacc[(rb + r) * HACC_STRIDE + 16 + colL] = d1[r];
        }
        __syncthreads();

        // Gather this lane's A-fragment columns, add embedding, tanh, -> f16.
        const float4* hrr = (const float4*)&hacc[row * HACC_STRIDE];
        const float4 h0 = hrr[c0i], h1 = hrr[c0i + 1];
        const float4 h2 = hrr[c1i], h3 = hrr[c1i + 1];

        float s[16];
        s[0]=h0.x+we0.x; s[1]=h0.y+we0.y; s[2]=h0.z+we0.z; s[3]=h0.w+we0.w;
        s[4]=h1.x+we1.x; s[5]=h1.y+we1.y; s[6]=h1.z+we1.z; s[7]=h1.w+we1.w;
        s[8]=h2.x+we2.x; s[9]=h2.y+we2.y; s[10]=h2.z+we2.z; s[11]=h2.w+we2.w;
        s[12]=h3.x+we3.x; s[13]=h3.y+we3.y; s[14]=h3.z+we3.z; s[15]=h3.w+we3.w;

        #pragma unroll
        for (int i = 0; i < 16; ++i)
            a[i] = (_Float16)fast_tanhf(s[i]);

        // Persist fragment: 32B contiguous per lane -> fully coalesced.
        v8h lo, hi;
        #pragma unroll
        for (int i = 0; i < 8; ++i) { lo[i] = a[i]; hi[i] = a[i + 8]; }
        _Float16* dst = hsFrag + ((size_t)(bg * T_ + t) * 32 + lane) * 16;
        *(v8h*)(dst)     = lo;
        *(v8h*)(dst + 8) = hi;

        __syncthreads();   // protect hacc reads from next iteration's scatter
    }
}

// ---------------------------------------------------------------------------
// Kernel 2: output projection  hs[BT,32] @ W_hq[32,256] + b_q  (store-bound).
// Each wave owns two resident B-fragments (32 columns of W_hq) with b_q
// pre-loaded into the WMMA C accumulator; streams A-fragments from workspace
// (L2-resident, written by kernel 1). Output stores are non-temporal: the
// 256 MB stream should not rinse hsFrag out of the 192 MB L2.
// ---------------------------------------------------------------------------
__global__ __launch_bounds__(256)
void rnn_logits_kernel(const _Float16* __restrict__ hsFrag, // [BG*T][32][16]
                       const float*    __restrict__ W_hq,   // [H, V]
                       const float*    __restrict__ b_q,    // [V]
                       float*          __restrict__ out)    // [B, T, V]
{
    const int lane = threadIdx.x & 31;
    const int wave = threadIdx.x >> 5;     // 0..7 -> column tiles 2w, 2w+1
    const int colL = lane & 15;
    const int kb   = (lane < 16) ? 0 : 8;
    const int nb   = wave * 32;

    // Resident B-fragments of W_hq for two 16-column tiles.
    v16h bA, bB;
    #pragma unroll
    for (int i = 0; i < 8; ++i) {
        const int k1 = kb + i;
        const int k2 = kb + 16 + i;
        bA[i]     = (_Float16)W_hq[k1 * V_ + nb + colL];
        bA[i + 8] = (_Float16)W_hq[k2 * V_ + nb + colL];
        bB[i]     = (_Float16)W_hq[k1 * V_ + nb + 16 + colL];
        bB[i + 8] = (_Float16)W_hq[k2 * V_ + nb + 16 + colL];
    }
    // Bias fused into the accumulator: C lane holds one column for 8 rows.
    v8f cA, cB;
    {
        const float qA = b_q[nb + colL];
        const float qB = b_q[nb + 16 + colL];
        #pragma unroll
        for (int i = 0; i < 8; ++i) { cA[i] = qA; cB[i] = qB; }
    }

    const int rb    = (lane < 16) ? 0 : 8;
    const int numRG = BG_ * T_;            // 16384 row-groups of 16 rows
    for (int rg = blockIdx.x; rg < numRG; rg += gridDim.x) {
        const int bg = rg >> 8;
        const int t  = rg & (T_ - 1);

        const v16h aval = *(const v16h*)(hsFrag + ((size_t)rg * 32 + lane) * 16);

        v8f dA = __builtin_amdgcn_wmma_f32_16x16x32_f16(false, aval, false, bA,
                                                        (short)0, cA, false, false);
        v8f dB = __builtin_amdgcn_wmma_f32_16x16x32_f16(false, aval, false, bB,
                                                        (short)0, cB, false, false);

        #pragma unroll
        for (int r = 0; r < 8; ++r) {
            const int b = bg * 16 + rb + r;
            const size_t o = ((size_t)(b * T_ + t)) * V_ + nb + colL;
            __builtin_nontemporal_store(dA[r], &out[o]);
            __builtin_nontemporal_store(dB[r], &out[o + 16]);
        }
    }
}

// ---------------------------------------------------------------------------
extern "C" void kernel_launch(void* const* d_in, const int* in_sizes, int n_in,
                              void* d_out, int out_size, void* d_ws, size_t ws_size,
                              hipStream_t stream) {
    (void)in_sizes; (void)n_in; (void)out_size; (void)ws_size;
    const int*   X    = (const int*)  d_in[0];
    const float* W_xh = (const float*)d_in[1];
    const float* W_hh = (const float*)d_in[2];
    const float* b_h  = (const float*)d_in[3];
    const float* W_hq = (const float*)d_in[4];
    const float* b_q  = (const float*)d_in[5];
    float*    out    = (float*)d_out;
    _Float16* hsFrag = (_Float16*)d_ws;   // needs BG*T*32*16*2 = 16.8 MB

    rnn_scan_kernel<<<BG_, 32, 0, stream>>>(X, W_xh, W_hh, b_h, hsFrag);
    rnn_logits_kernel<<<512, 256, 0, stream>>>(hsFrag, W_hq, b_q, out);
}
